// Self_attention_10720238370834
// MI455X (gfx1250) — matrix operand order repaired and checked by reading, not compiled, not run on hardware
//
#include <hip/hip_runtime.h>

// ---------------------------------------------------------------------------
// Self-attention with query-axis softmax, CDNA5 (gfx1250) WMMA bf16 pipeline.
//   B=4, N=4096, D=512
//   Q=XWq^T+bq ; K=XWk^T+bk ; V=XWv^T+bv
//   S=Q K^T ; P=softmax(S, axis=q) ; O=P V
// S (256 MiB) is never materialized: streaming column stats + recompute.
// W pre-converted to bf16; BM=32 tiles so each B-fragment feeds 2 WMMAs.
// ---------------------------------------------------------------------------

typedef __attribute__((ext_vector_type(16))) __bf16 v16bf;
typedef __attribute__((ext_vector_type(8)))  float  v8f;

#define BATCH 4
#define SEQ   4096
#define DIM   512
#define MTOT  (BATCH * SEQ)

// hardware f32 -> bf16 (lowers to v_cvt_*bf16_f32)
static __device__ inline unsigned short f2bf(float f) {
  return __builtin_bit_cast(unsigned short, (__bf16)f);
}
static __device__ inline unsigned int pk2(float a, float b) {
  return (unsigned int)f2bf(a) | ((unsigned int)f2bf(b) << 16);
}

struct U16x16 { uint4 lo, hi; };            // 32 bytes == v16bf

// two 16-byte chunks -> one bf16 fragment (elements 0..7 = p0, 8..15 = p1)
static __device__ inline v16bf ldfrag(const unsigned short* p0,
                                      const unsigned short* p1) {
  U16x16 t;
  t.lo = *(const uint4*)p0;
  t.hi = *(const uint4*)p1;
  return __builtin_bit_cast(v16bf, t);
}

static __device__ inline v8f zero8() {
  v8f z = {0.f, 0.f, 0.f, 0.f, 0.f, 0.f, 0.f, 0.f};
  return z;
}

static __device__ inline v8f wmma_bf16(v16bf a, v16bf b, v8f c) {
  return __builtin_amdgcn_wmma_f32_16x16x32_bf16(
      /*neg_a=*/false, a, /*neg_b=*/false, b,
      /*c_mod=*/(short)0, c, /*reuse_a=*/false, /*reuse_b=*/false);
}

// ---------------------------------------------------------------------------
// Kernel 0: one-shot weight conversion f32 -> bf16.
// ---------------------------------------------------------------------------
__global__ __launch_bounds__(256) void convert_w_kernel(
    const float* __restrict__ Wq, const float* __restrict__ Wk,
    const float* __restrict__ Wv, unsigned short* __restrict__ Wb) {
  const int mat = blockIdx.y;
  const float* W = (mat == 0) ? Wq : (mat == 1) ? Wk : Wv;
  const int idx = (blockIdx.x * 256 + threadIdx.x) * 4;   // < 512*512
  float4 x = *(const float4*)&W[idx];
  *(uint2*)&Wb[(size_t)mat * DIM * DIM + idx] =
      make_uint2(pk2(x.x, x.y), pk2(x.z, x.w));
}

// ---------------------------------------------------------------------------
// Kernel 1: fused QKV projection, BM=32.  Y[n,e] = sum_d X[n,d]*W[e,d] + b[e]
//   Each W B-fragment is reused by two A-fragments (rows 0-15 / 16-31).
//   mode 0 -> Qb row-major, 1 -> Kb row-major, 2 -> Vt transposed [b][e][n].
// grid (M/32, 3), 256 threads (8 waves, 64 output cols per wave).
// ---------------------------------------------------------------------------
__global__ __launch_bounds__(256) void qkv_proj_kernel(
    const float* __restrict__ X, const unsigned short* __restrict__ Wb,
    const float* __restrict__ bq, const float* __restrict__ bk,
    const float* __restrict__ bv,
    unsigned short* __restrict__ Qb, unsigned short* __restrict__ Kb,
    unsigned short* __restrict__ Vt) {
  __shared__ unsigned short Xs[32][520];    // 32x512 bf16, padded rows

  const int tid   = threadIdx.x;
  const int wave  = tid >> 5;
  const int lane  = tid & 31;
  const int lhalf = lane >> 4;
  const int lcol  = lane & 15;
  const int mode  = blockIdx.y;
  const int m0    = blockIdx.x * 32;

  const unsigned short* W = Wb + (size_t)mode * DIM * DIM;
  const float* bias = (mode == 0) ? bq : (mode == 1) ? bk : bv;

  // stage X tile, converting f32 -> bf16 (hardware cvt)
  for (int i = tid; i < 32 * DIM / 4; i += 256) {
    int row = (i * 4) >> 9, col = (i * 4) & 511;
    float4 x = *(const float4*)&X[(size_t)(m0 + row) * DIM + col];
    *(uint2*)&Xs[row][col] = make_uint2(pk2(x.x, x.y), pk2(x.z, x.w));
  }
  __syncthreads();

  v8f acc0[4] = {zero8(), zero8(), zero8(), zero8()};   // rows m0+0..15
  v8f acc1[4] = {zero8(), zero8(), zero8(), zero8()};   // rows m0+16..31

  for (int dk = 0; dk < 16; ++dk) {
    const int d = dk * 32;
    const unsigned short* a0 = &Xs[lcol][d + lhalf * 8];
    const unsigned short* a1 = &Xs[16 + lcol][d + lhalf * 8];
    v16bf af0 = ldfrag(a0, a0 + 16);
    v16bf af1 = ldfrag(a1, a1 + 16);
#pragma unroll
    for (int nt = 0; nt < 4; ++nt) {
      const int col = wave * 64 + nt * 16 + lcol;
      const unsigned short* wp = &W[(size_t)col * DIM + d + lhalf * 8];
      v16bf bf = ldfrag(wp, wp + 16);
      acc0[nt] = wmma_bf16(af0, bf, acc0[nt]);
      acc1[nt] = wmma_bf16(af1, bf, acc1[nt]);
    }
  }

  const int batch = m0 >> 12;
  const int nIdx  = m0 & (SEQ - 1);
#pragma unroll
  for (int nt = 0; nt < 4; ++nt) {
    const int col = wave * 64 + nt * 16 + lcol;
    const float bb = bias[col];
    if (mode < 2) {
      unsigned short* Out = (mode == 0) ? Qb : Kb;
#pragma unroll
      for (int r = 0; r < 8; ++r) {
        const int rowM = r + lhalf * 8;
        Out[(size_t)(m0 + rowM) * DIM + col]      = f2bf(acc0[nt][r] + bb);
        Out[(size_t)(m0 + 16 + rowM) * DIM + col] = f2bf(acc1[nt][r] + bb);
      }
    } else {
      // transposed store: element (M,N) -> Vt[batch][N][nIdx+M]; 8 consecutive M
      unsigned short* vr = &Vt[((size_t)batch * DIM + col) * SEQ + nIdx];
      uint4 p0 = make_uint4(pk2(acc0[nt][0] + bb, acc0[nt][1] + bb),
                            pk2(acc0[nt][2] + bb, acc0[nt][3] + bb),
                            pk2(acc0[nt][4] + bb, acc0[nt][5] + bb),
                            pk2(acc0[nt][6] + bb, acc0[nt][7] + bb));
      uint4 p1 = make_uint4(pk2(acc1[nt][0] + bb, acc1[nt][1] + bb),
                            pk2(acc1[nt][2] + bb, acc1[nt][3] + bb),
                            pk2(acc1[nt][4] + bb, acc1[nt][5] + bb),
                            pk2(acc1[nt][6] + bb, acc1[nt][7] + bb));
      *(uint4*)&vr[lhalf * 8]      = p0;
      *(uint4*)&vr[16 + lhalf * 8] = p1;
    }
  }
}

// ---------------------------------------------------------------------------
// Kernel 2: per-key-column streaming softmax stats over the QUERY axis.
//   m_k = max_q S[q,k]; s_k = sum_q exp(S[q,k]-m_k); online rescaled update.
// grid (N/128, B), 256 threads; each wave owns 16 key columns, K-fragments
// register-hoisted, loops over all 256 q-tiles recomputing S via WMMA.
// ---------------------------------------------------------------------------
__global__ __launch_bounds__(256) void softmax_stats_kernel(
    const unsigned short* __restrict__ Qb, const unsigned short* __restrict__ Kb,
    float* __restrict__ colmax, float* __restrict__ colsum) {
  const int tid   = threadIdx.x;
  const int wave  = tid >> 5;
  const int lane  = tid & 31;
  const int lhalf = lane >> 4;
  const int lcol  = lane & 15;
  const int batch = blockIdx.y;
  const int kcol  = blockIdx.x * 128 + wave * 16 + lcol;

  // hoist this wave's K block fragments (16 rows x 512) into registers
  const size_t krow = ((size_t)batch * SEQ + kcol) * DIM;
  v16bf kf[16];
#pragma unroll
  for (int dk = 0; dk < 16; ++dk) {
    const unsigned short* p = &Kb[krow + dk * 32 + lhalf * 8];
    kf[dk] = ldfrag(p, p + 16);
  }

  float m = -3.0e38f, s = 0.0f;
  for (int qt = 0; qt < SEQ / 16; ++qt) {
    v8f c = zero8();
#pragma unroll
    for (int dk = 0; dk < 16; ++dk) {
      const unsigned short* ap =
          &Qb[((size_t)batch * SEQ + qt * 16 + lcol) * DIM + dk * 32 + lhalf * 8];
      c = wmma_bf16(ldfrag(ap, ap + 16), kf[dk], c);
    }
    // column reduce: 8 in-lane rows + partner lane (lane ^ 16) holds same col
    float tmax = c[0];
#pragma unroll
    for (int r = 1; r < 8; ++r) tmax = fmaxf(tmax, c[r]);
    tmax = fmaxf(tmax, __shfl_xor(tmax, 16, 32));
    const float nm = fmaxf(m, tmax);
    float ls = 0.f;
#pragma unroll
    for (int r = 0; r < 8; ++r) ls += __expf(c[r] - nm);
    ls += __shfl_xor(ls, 16, 32);
    s = s * __expf(m - nm) + ls;
    m = nm;
  }
  if (lane < 16) {
    colmax[(size_t)batch * SEQ + kcol] = m;
    colsum[(size_t)batch * SEQ + kcol] = s;
  }
}

// ---------------------------------------------------------------------------
// Kernel 3: O = softmax(S, axis=q) @ V, S recomputed tile-by-tile.  BM=32.
// grid (N/32, B), 256 threads. Per WG: 32 q rows x all 512 d cols.
// Each k-step (128 keys): every wave recomputes two 16x16 S subtiles sharing
// one Kb fragment, applies exp((S-m_k))/s_k, round-trips P through LDS
// (C-frag -> A-frag), then accumulates P @ V^T-rows with WMMA, each Vt
// fragment shared by two P A-fragments.
// ---------------------------------------------------------------------------
__global__ __launch_bounds__(256) void attn_out_kernel(
    const unsigned short* __restrict__ Qb, const unsigned short* __restrict__ Kb,
    const unsigned short* __restrict__ Vt,
    const float* __restrict__ colmax, const float* __restrict__ colsum,
    float* __restrict__ O) {
  __shared__ unsigned short Qs[32][520];   // q tile, bf16
  __shared__ unsigned short Ps[32][136];   // probability tile [32 q][128 k]

  const int tid   = threadIdx.x;
  const int wave  = tid >> 5;
  const int lane  = tid & 31;
  const int lhalf = lane >> 4;
  const int lcol  = lane & 15;
  const int batch = blockIdx.y;
  const int q0    = blockIdx.x * 32;

  // stage Q tile (already bf16)
  for (int i = tid; i < 32 * DIM / 8; i += 256) {
    int row = (i * 8) >> 9, col = (i * 8) & 511;
    uint4 v = *(const uint4*)&Qb[((size_t)batch * SEQ + q0 + row) * DIM + col];
    *(uint4*)&Qs[row][col] = v;
  }
  __syncthreads();

  v8f oacc0[4] = {zero8(), zero8(), zero8(), zero8()};   // q rows 0..15
  v8f oacc1[4] = {zero8(), zero8(), zero8(), zero8()};   // q rows 16..31

  for (int kt = 0; kt < SEQ / 128; ++kt) {
    const int k0   = kt * 128;
    const int kcol = k0 + wave * 16 + lcol;

    // prefetch next k-step working set into cache (global_prefetch)
    if (kt + 1 < SEQ / 128) {
      const int nkcol = kcol + 128;
      __builtin_prefetch(&Kb[((size_t)batch * SEQ + nkcol) * DIM + lhalf * 256],
                         0, 1);
      __builtin_prefetch(
          &Vt[((size_t)batch * DIM + wave * 64 + lcol * 4) * SEQ + k0 + 128 +
              lhalf * 64],
          0, 1);
    }

    // ---- phase A: recompute two S subtiles for this wave's 16 key cols ----
    v8f c0 = zero8(), c1 = zero8();
    const size_t krow = ((size_t)batch * SEQ + kcol) * DIM;
#pragma unroll
    for (int dk = 0; dk < 16; ++dk) {
      const unsigned short* a0 = &Qs[lcol][dk * 32 + lhalf * 8];
      const unsigned short* a1 = &Qs[16 + lcol][dk * 32 + lhalf * 8];
      const unsigned short* bp = &Kb[krow + dk * 32 + lhalf * 8];
      v16bf bf = ldfrag(bp, bp + 16);
      c0 = wmma_bf16(ldfrag(a0, a0 + 16), bf, c0);
      c1 = wmma_bf16(ldfrag(a1, a1 + 16), bf, c1);
    }
    const float mk = colmax[(size_t)batch * SEQ + kcol];
    const float rs = 1.0f / colsum[(size_t)batch * SEQ + kcol];
    const int  pc = wave * 16 + lcol;
#pragma unroll
    for (int r = 0; r < 8; ++r) {
      Ps[r + lhalf * 8][pc]      = f2bf(__expf(c0[r] - mk) * rs);
      Ps[16 + r + lhalf * 8][pc] = f2bf(__expf(c1[r] - mk) * rs);
    }
    __syncthreads();

    // ---- phase B: O += P @ V  (V^T rows are contiguous in Vt) ----
#pragma unroll
    for (int ks = 0; ks < 4; ++ks) {
      const unsigned short* a0 = &Ps[lcol][ks * 32 + lhalf * 8];
      const unsigned short* a1 = &Ps[16 + lcol][ks * 32 + lhalf * 8];
      v16bf af0 = ldfrag(a0, a0 + 16);
      v16bf af1 = ldfrag(a1, a1 + 16);
#pragma unroll
      for (int nt = 0; nt < 4; ++nt) {
        const int dcol = wave * 64 + nt * 16 + lcol;
        const unsigned short* vp =
            &Vt[((size_t)batch * DIM + dcol) * SEQ + k0 + ks * 32 + lhalf * 8];
        v16bf bf = ldfrag(vp, vp + 16);
        oacc0[nt] = wmma_bf16(af0, bf, oacc0[nt]);
        oacc1[nt] = wmma_bf16(af1, bf, oacc1[nt]);
      }
    }
    __syncthreads();
  }

  // store O (f32)
#pragma unroll
  for (int nt = 0; nt < 4; ++nt) {
    const int col = wave * 64 + nt * 16 + lcol;
#pragma unroll
    for (int r = 0; r < 8; ++r) {
      const int rowM = r + lhalf * 8;
      O[((size_t)batch * SEQ + q0 + rowM) * DIM + col]      = oacc0[nt][r];
      O[((size_t)batch * SEQ + q0 + 16 + rowM) * DIM + col] = oacc1[nt][r];
    }
  }
}

// ---------------------------------------------------------------------------
extern "C" void kernel_launch(void* const* d_in, const int* in_sizes, int n_in,
                              void* d_out, int out_size, void* d_ws, size_t ws_size,
                              hipStream_t stream) {
  const float* X  = (const float*)d_in[0];
  const float* Wq = (const float*)d_in[1];
  const float* bq = (const float*)d_in[2];
  const float* Wk = (const float*)d_in[3];
  const float* bk = (const float*)d_in[4];
  const float* Wv = (const float*)d_in[5];
  const float* bv = (const float*)d_in[6];
  float* O = (float*)d_out;

  // workspace: Qb | Kb | Vt (bf16) | colmax | colsum | Wb   (~49.6 MiB)
  unsigned short* Qb = (unsigned short*)d_ws;
  unsigned short* Kb = Qb + (size_t)MTOT * DIM;
  unsigned short* Vt = Kb + (size_t)MTOT * DIM;
  float* colmax = (float*)(Vt + (size_t)MTOT * DIM);
  float* colsum = colmax + MTOT;
  unsigned short* Wb = (unsigned short*)(colsum + MTOT);

  convert_w_kernel<<<dim3(DIM * DIM / (256 * 4), 3), 256, 0, stream>>>(
      Wq, Wk, Wv, Wb);
  qkv_proj_kernel<<<dim3(MTOT / 32, 3), 256, 0, stream>>>(
      X, Wb, bq, bk, bv, Qb, Kb, Vt);
  softmax_stats_kernel<<<dim3(SEQ / 128, BATCH), 256, 0, stream>>>(
      Qb, Kb, colmax, colsum);
  attn_out_kernel<<<dim3(SEQ / 32, BATCH), 256, 0, stream>>>(
      Qb, Kb, Vt, colmax, colsum, O);
}
